// DSF_GPR_I_61357902790938
// MI455X (gfx1250) — matrix operand order repaired and checked by reading, not compiled, not run on hardware
//
#include <hip/hip_runtime.h>

#define NN 6000
#define NPAD 6016
#define EE 192000
#define IND 512
#define HIDD 256
#define OUTD 64
#define PEIN 32
#define PEH 64
#define KIT 10

typedef __attribute__((ext_vector_type(16))) __bf16 v16bf;
typedef __attribute__((ext_vector_type(8)))  __bf16 v8bf;
typedef __attribute__((ext_vector_type(8)))  float  v8f;

static __device__ __forceinline__ __bf16 f2bf(float f) {
  unsigned u = __builtin_bit_cast(unsigned, f);
  unsigned r = (u + 0x7FFFu + ((u >> 16) & 1u)) >> 16;   // round-to-nearest-even
  unsigned short s = (unsigned short)r;
  return __builtin_bit_cast(__bf16, s);
}

// Fast sigmoid: v_exp_f32 + v_rcp_f32 (no IEEE div sequence). ~1ulp rcp is far
// below the bf16 quantization of the consumer.
static __device__ __forceinline__ float fsigmoid(float x) {
  return __builtin_amdgcn_rcpf(1.0f + __expf(-x));
}

// ---------------- CSR build ----------------
__global__ void gpr_zeroi(int* p, int n) {
  int i = blockIdx.x * blockDim.x + threadIdx.x;
  if (i < n) p[i] = 0;
}
__global__ void gpr_count(const int* __restrict__ col, int* __restrict__ cnt) {
  int e = blockIdx.x * blockDim.x + threadIdx.x;
  if (e < EE) atomicAdd(&cnt[col[e]], 1);
}
__global__ void gpr_dinv(const int* __restrict__ cnt, float* __restrict__ dinv) {
  int i = blockIdx.x * blockDim.x + threadIdx.x;
  if (i < NN) dinv[i] = rsqrtf((float)(cnt[i] + 1));   // +1 self loop; deg >= 1 always
}
__global__ __launch_bounds__(256) void gpr_scan(const int* __restrict__ cnt,
                                                int* __restrict__ roff,
                                                int* __restrict__ cursor) {
  __shared__ int sums[256];
  const int t = threadIdx.x;
  const int base = t * 24;                 // 256*24 = 6144 >= 6000
  int s = 0;
#pragma unroll 4
  for (int i = 0; i < 24; ++i) {
    int idx = base + i;
    if (idx < NN) s += cnt[idx];
  }
  sums[t] = s;
  __syncthreads();
  for (int off = 1; off < 256; off <<= 1) {   // inclusive Hillis-Steele scan
    int add = (t >= off) ? sums[t - off] : 0;
    __syncthreads();
    sums[t] += add;
    __syncthreads();
  }
  int run = (t == 0) ? 0 : sums[t - 1];
  for (int i = 0; i < 24; ++i) {
    int idx = base + i;
    if (idx < NN) {
      roff[idx] = run;
      cursor[idx] = run;
      run += cnt[idx];
    }
  }
  if (t == 255) roff[NN] = sums[255];
}
__global__ void gpr_fill(const int* __restrict__ row, const int* __restrict__ col,
                         int* __restrict__ cursor, int* __restrict__ csr) {
  int e = blockIdx.x * blockDim.x + threadIdx.x;
  if (e < EE) {
    int p = atomicAdd(&cursor[col[e]], 1);
    csr[p] = row[e];
  }
}

// ---------------- MLP: x = relu(nf@W1+b1)@W2+b2 ----------------
__global__ __launch_bounds__(256) void gpr_mlp(const float* __restrict__ nf,
    const float* __restrict__ W1, const float* __restrict__ b1,
    const float* __restrict__ W2, const float* __restrict__ b2,
    float* __restrict__ x) {
  __shared__ float hid[HIDD];
  const int n = blockIdx.x;
  const int t = threadIdx.x;
  const float* row = nf + (size_t)n * IND;
  float acc = b1[t];
#pragma unroll 4
  for (int k = 0; k < IND; ++k) acc += row[k] * W1[(size_t)k * HIDD + t];
  hid[t] = fmaxf(acc, 0.f);
  __syncthreads();
  if (t < OUTD) {
    float a2 = b2[t];
#pragma unroll 4
    for (int j = 0; j < HIDD; ++j) a2 += hid[j] * W2[(size_t)j * OUTD + t];
    x[(size_t)n * OUTD + t] = a2;
  }
}

// ---------------- pe = tanh(pos@Wpe+bpe) ----------------
__global__ __launch_bounds__(64) void gpr_peinit(const float* __restrict__ pos,
    const float* __restrict__ Wpe, const float* __restrict__ bpe,
    float* __restrict__ pe, float* __restrict__ raw) {
  const int n = blockIdx.x, j = threadIdx.x;
  float acc = bpe[j];
#pragma unroll
  for (int k = 0; k < PEIN; ++k) acc += pos[(size_t)n * PEIN + k] * Wpe[(size_t)k * PEH + j];
  float v = tanhf(acc);
  pe[(size_t)n * PEH + j] = v;
  raw[(size_t)n * PEH + j] = v;
}

// ---------------- hidden = x * gamma0 ----------------
__global__ __launch_bounds__(64) void gpr_gamma0(const float* __restrict__ pe,
    const float* __restrict__ x, const float* __restrict__ temp,
    const float* __restrict__ cw, const float* __restrict__ cb,
    float* __restrict__ hidden) {
  __shared__ float red[64];
  const int n = blockIdx.x, c = threadIdx.x;
  red[c] = pe[(size_t)n * 64 + c] * cw[c];
  __syncthreads();
  for (int s2 = 32; s2 > 0; s2 >>= 1) {
    if (c < s2) red[c] += red[c + s2];
    __syncthreads();
  }
  float g = temp[0] * tanhf(red[0] + cb[0]);
  hidden[(size_t)n * 64 + c] = x[(size_t)n * 64 + c] * g;
}

// ---------------- q = pe@Wc+bc (bf16), pT = pe^T (bf16), zero-padded ----------------
__global__ __launch_bounds__(64) void gpr_qpt(const float* __restrict__ pe,
    const float* __restrict__ Wc, const float* __restrict__ bc,
    __bf16* __restrict__ q, __bf16* __restrict__ pT) {
  const int n = blockIdx.x, j = threadIdx.x;
  if (n >= NN) {                       // whole block uniform: no barrier divergence
    q[(size_t)n * PEH + j] = f2bf(0.f);
    pT[(size_t)j * NPAD + n] = f2bf(0.f);
    return;
  }
  __shared__ float row[PEH];
  row[j] = pe[(size_t)n * PEH + j];
  __syncthreads();
  float acc = bc[j];
#pragma unroll 4
  for (int c = 0; c < PEH; ++c) acc += row[c] * Wc[(size_t)c * PEH + j];
  q[(size_t)n * PEH + j] = f2bf(acc);
  pT[(size_t)j * NPAD + n] = f2bf(row[j]);
}

// ---------------- gather prop for x and pe (includes self loop) ----------------
__global__ __launch_bounds__(64) void gpr_prop(const int* __restrict__ roff,
    const int* __restrict__ csr, const float* __restrict__ dinv,
    const float* __restrict__ x, const float* __restrict__ pe,
    float* __restrict__ xo, float* __restrict__ po) {
  const int n = blockIdx.x;
  const int f = threadIdx.x;
  const float di = dinv[n];
  const int s = roff[n], e = roff[n + 1];
  float ax = di * di * x[(size_t)n * 64 + f];
  float ap = di * di * pe[(size_t)n * 64 + f];
  for (int i = s; i < e; ++i) {
    const int src = csr[i];
    const float w = di * dinv[src];
    ax += w * x[(size_t)src * 64 + f];
    ap += w * pe[(size_t)src * 64 + f];
  }
  xo[(size_t)n * 64 + f] = ax;
  po[(size_t)n * 64 + f] = ap;
}

// ---------------- fused corr: out = sigmoid(q q^T) @ p  (flash-style, WMMA bf16) ----
// q:  [NPAD][64] bf16 row-major (pad rows zero)
// pT: [64][NPAD] bf16 (pad cols zero)
// Block = 8 waves; each wave owns a j-slice; LDS reduction at the end.
__global__ __launch_bounds__(256) void gpr_corr(const __bf16* __restrict__ q,
                                                const __bf16* __restrict__ pT,
                                                float* __restrict__ out) {
  __shared__ float red[8 * 16 * 64];           // 32 KB
  const int m0 = blockIdx.x << 4;
  const int lane = threadIdx.x & 31;
  const int wave = threadIdx.x >> 5;
  const int r = lane & 15;
  const int h = lane >> 4;

  // Stage-1 B operand: q_i rows m0..m0+15 in B layout (col n in lanes n, n+16;
  // K 0..15 in lanes<16, K 16..31 in lanes>=16, contiguous in row-major memory).
  const __bf16* qi = q + (size_t)(m0 + r) * PEH;
  const v16bf bi0 = *reinterpret_cast<const v16bf*>(qi + 16 * h);        // K 0..31
  const v16bf bi1 = *reinterpret_cast<const v16bf*>(qi + 32 + 16 * h);   // K 32..63

  v8f acc0, acc1, acc2, acc3;
#pragma unroll
  for (int t = 0; t < 8; ++t) { acc0[t] = 0.f; acc1[t] = 0.f; acc2[t] = 0.f; acc3[t] = 0.f; }

  for (int jj = wave; jj < NPAD / 32; jj += 8) {
    const int j0 = jj << 5;
    const __bf16* qa = q + (size_t)(j0 + r + (r & 8)) * PEH;        // j-tile rows j0..j0+7, j0+16..j0+23
    const __bf16* qb = q + (size_t)(j0 + 8 + r + (r & 8)) * PEH;    // j-tile rows j0+8..j0+15, j0+24..j0+31
    v8f d1, d2;
#pragma unroll
    for (int t = 0; t < 8; ++t) { d1[t] = 0.f; d2[t] = 0.f; }

    v16bf a; v8bf lo, hi;
    // A operand (16x32 bf16), contiguous k order chosen here to match B: lane<16 holds K k0..k0+15; lane>=16 holds K k0+16..k0+31.
    lo = *reinterpret_cast<const v8bf*>(qa + 16 * h);
    hi = *reinterpret_cast<const v8bf*>(qa + 8 + 16 * h);
#pragma unroll
    for (int t = 0; t < 8; ++t) { a[t] = lo[t]; a[8 + t] = hi[t]; }
    d1 = __builtin_amdgcn_wmma_f32_16x16x32_bf16(false, a, false, bi0, (short)0, d1, false, false);
    lo = *reinterpret_cast<const v8bf*>(qa + 32 + 16 * h);
    hi = *reinterpret_cast<const v8bf*>(qa + 40 + 16 * h);
#pragma unroll
    for (int t = 0; t < 8; ++t) { a[t] = lo[t]; a[8 + t] = hi[t]; }
    d1 = __builtin_amdgcn_wmma_f32_16x16x32_bf16(false, a, false, bi1, (short)0, d1, false, false);

    lo = *reinterpret_cast<const v8bf*>(qb + 16 * h);
    hi = *reinterpret_cast<const v8bf*>(qb + 8 + 16 * h);
#pragma unroll
    for (int t = 0; t < 8; ++t) { a[t] = lo[t]; a[8 + t] = hi[t]; }
    d2 = __builtin_amdgcn_wmma_f32_16x16x32_bf16(false, a, false, bi0, (short)0, d2, false, false);
    lo = *reinterpret_cast<const v8bf*>(qb + 32 + 16 * h);
    hi = *reinterpret_cast<const v8bf*>(qb + 40 + 16 * h);
#pragma unroll
    for (int t = 0; t < 8; ++t) { a[t] = lo[t]; a[8 + t] = hi[t]; }
    d2 = __builtin_amdgcn_wmma_f32_16x16x32_bf16(false, a, false, bi1, (short)0, d2, false, false);

    // D (= transposed logits) layout coincides with the next A layout lane-for-lane:
    // lane (r,h) holds L[r][16h+v] in d1[v] and L[r][16h+8+v] in d2[v].
    v16bf a2;
#pragma unroll
    for (int t = 0; t < 8; ++t) {
      a2[t]     = f2bf(fsigmoid(d1[t]));
      a2[8 + t] = f2bf(fsigmoid(d2[t]));
    }
    // Stage-2 B operand: p columns from pT (contiguous along j).
    const __bf16* pc = pT + (size_t)r * NPAD + j0 + 16 * h;
    acc0 = __builtin_amdgcn_wmma_f32_16x16x32_bf16(false, a2, false,
             *reinterpret_cast<const v16bf*>(pc), (short)0, acc0, false, false);
    acc1 = __builtin_amdgcn_wmma_f32_16x16x32_bf16(false, a2, false,
             *reinterpret_cast<const v16bf*>(pc + (size_t)16 * NPAD), (short)0, acc1, false, false);
    acc2 = __builtin_amdgcn_wmma_f32_16x16x32_bf16(false, a2, false,
             *reinterpret_cast<const v16bf*>(pc + (size_t)32 * NPAD), (short)0, acc2, false, false);
    acc3 = __builtin_amdgcn_wmma_f32_16x16x32_bf16(false, a2, false,
             *reinterpret_cast<const v16bf*>(pc + (size_t)48 * NPAD), (short)0, acc3, false, false);
  }

  // Reduce the 8 per-wave partial 16x64 tiles through LDS.
  float* myred = red + wave * 1024;
#pragma unroll
  for (int t = 0; t < 8; ++t) {
    int m = t + 8 * h;                       // C/D layout: VGPR v <-> M = v + 8*(lane>>4)
    myred[m * 64 + r]      = acc0[t];
    myred[m * 64 + 16 + r] = acc1[t];
    myred[m * 64 + 32 + r] = acc2[t];
    myred[m * 64 + 48 + r] = acc3[t];
  }
  __syncthreads();
  const int tid = threadIdx.x;
#pragma unroll
  for (int u = 0; u < 4; ++u) {
    int pos = tid * 4 + u;
    float s = 0.f;
#pragma unroll
    for (int w = 0; w < 8; ++w) s += red[w * 1024 + pos];
    out[(size_t)m0 * 64 + pos] = s;
  }
}

// ---------------- pe update ----------------
__global__ void gpr_peup(const float* __restrict__ tpo, const float* __restrict__ corr,
                         const float* __restrict__ raw, float* __restrict__ pe, int total) {
  int i = blockIdx.x * blockDim.x + threadIdx.x;
  if (i < total) {
    float v = 1.5f * tpo[i] - 0.5f * corr[i];     // (1+beta)*tpo - beta*corr
    pe[i] = tanhf(0.5f * raw[i] + 0.5f * v);      // alpha = 0.5
  }
}

// ---------------- hidden += gamma_{k+1} * x_new ----------------
__global__ __launch_bounds__(64) void gpr_hx(const float* __restrict__ pe,
    const float* __restrict__ xnew, const float* __restrict__ temp,
    const float* __restrict__ cw, const float* __restrict__ cb,
    float* __restrict__ hidden, int k1) {
  __shared__ float red[64];
  const int n = blockIdx.x, c = threadIdx.x;
  red[c] = pe[(size_t)n * 64 + c] * cw[(size_t)k1 * 64 + c];
  __syncthreads();
  for (int s2 = 32; s2 > 0; s2 >>= 1) {
    if (c < s2) red[c] += red[c + s2];
    __syncthreads();
  }
  float g = temp[k1] * tanhf(red[0] + cb[k1]);
  hidden[(size_t)n * 64 + c] += g * xnew[(size_t)n * 64 + c];
}

__global__ void gpr_copy(const float* __restrict__ src, float* __restrict__ dst, int total) {
  int i = blockIdx.x * blockDim.x + threadIdx.x;
  if (i < total) dst[i] = src[i];
}

extern "C" void kernel_launch(void* const* d_in, const int* in_sizes, int n_in,
                              void* d_out, int out_size, void* d_ws, size_t ws_size,
                              hipStream_t stream) {
  (void)in_sizes; (void)n_in; (void)out_size; (void)ws_size;
  const float* node_feat = (const float*)d_in[0];
  const float* pos_enc   = (const float*)d_in[1];
  const int*   edge      = (const int*)d_in[2];
  const float* W1  = (const float*)d_in[3];
  const float* b1  = (const float*)d_in[4];
  const float* W2  = (const float*)d_in[5];
  const float* b2  = (const float*)d_in[6];
  const float* Wpe = (const float*)d_in[7];
  const float* bpe = (const float*)d_in[8];
  const float* Wc  = (const float*)d_in[9];
  const float* bc  = (const float*)d_in[10];
  const float* temp    = (const float*)d_in[11];
  const float* coeff_w = (const float*)d_in[12];
  const float* coeff_b = (const float*)d_in[13];
  const int* erow = edge;
  const int* ecol = edge + EE;

  const int TOT = NN * 64;
  size_t off = 0;
  char* base = (char*)d_ws;
  auto ALC = [&](size_t bytes) -> void* {
    void* p = base + off;
    off = (off + bytes + 255) & ~(size_t)255;
    return p;
  };
  float* dinv   = (float*)ALC((size_t)NN * 4);
  int*   cnt    = (int*)  ALC((size_t)NN * 4);
  int*   roff   = (int*)  ALC((size_t)(NN + 1) * 4);
  int*   cursor = (int*)  ALC((size_t)NN * 4);
  int*   csr    = (int*)  ALC((size_t)EE * 4);
  float* xA     = (float*)ALC((size_t)TOT * 4);
  float* xB     = (float*)ALC((size_t)TOT * 4);
  float* pe     = (float*)ALC((size_t)TOT * 4);
  float* raw    = (float*)ALC((size_t)TOT * 4);
  float* tpo    = (float*)ALC((size_t)TOT * 4);
  float* corr   = (float*)ALC((size_t)TOT * 4);
  __bf16* qbf   = (__bf16*)ALC((size_t)NPAD * 64 * 2);
  __bf16* pTbf  = (__bf16*)ALC((size_t)64 * NPAD * 2);
  float* hidden = (float*)d_out;
  float* peout  = hidden + TOT;

  // graph normalization + CSR build (once)
  gpr_zeroi<<<(NN + 255) / 256, 256, 0, stream>>>(cnt, NN);
  gpr_count<<<(EE + 255) / 256, 256, 0, stream>>>(ecol, cnt);
  gpr_dinv <<<(NN + 255) / 256, 256, 0, stream>>>(cnt, dinv);
  gpr_scan <<<1, 256, 0, stream>>>(cnt, roff, cursor);
  gpr_fill <<<(EE + 255) / 256, 256, 0, stream>>>(erow, ecol, cursor, csr);

  // encoders
  gpr_mlp   <<<NN, 256, 0, stream>>>(node_feat, W1, b1, W2, b2, xA);
  gpr_peinit<<<NN, 64, 0, stream>>>(pos_enc, Wpe, bpe, pe, raw);
  gpr_gamma0<<<NN, 64, 0, stream>>>(pe, xA, temp, coeff_w, coeff_b, hidden);

  float* xc = xA;
  float* xn = xB;
  for (int k = 0; k < KIT; ++k) {
    gpr_qpt <<<NPAD, 64, 0, stream>>>(pe, Wc, bc, qbf, pTbf);
    gpr_prop<<<NN, 64, 0, stream>>>(roff, csr, dinv, xc, pe, xn, tpo);
    gpr_corr<<<NN / 16, 256, 0, stream>>>(qbf, pTbf, corr);
    gpr_peup<<<(TOT + 255) / 256, 256, 0, stream>>>(tpo, corr, raw, pe, TOT);
    gpr_hx  <<<NN, 64, 0, stream>>>(pe, xn, temp, coeff_w, coeff_b, hidden, k + 1);
    float* t = xc; xc = xn; xn = t;
  }
  gpr_copy<<<(TOT + 255) / 256, 256, 0, stream>>>(pe, peout, TOT);
}
